// ContextQueryAttention_49143015801172
// MI455X (gfx1250) — matrix operand order repaired and checked by reading, not compiled, not run on hardware
//
#include <hip/hip_runtime.h>

typedef __bf16 bf16_t;
typedef bf16_t v16bf __attribute__((ext_vector_type(16)));
typedef bf16_t v8bf  __attribute__((ext_vector_type(8)));
typedef float  v8f   __attribute__((ext_vector_type(8)));

#define DEVINL __device__ __forceinline__

constexpr int B = 64, C = 1024, Q = 256, D = 128;
constexpr float NEGV = -1e30f;

DEVINL bf16_t f2bf(float x) { return (bf16_t)x; }

// A-matrix fragment (16x32 bf16 tile): lane (m = l&15, half = l>>4) holds row
// row0+m, K = k0 + 8*half + {0..7} and k0 + 16 + 8*half + {0..7}.
DEVINL v16bf load_a_frag(const bf16_t* base, int stride, int row0, int k0, int lane) {
  int m = lane & 15, h = lane >> 4;
  const bf16_t* p = base + (size_t)(row0 + m) * stride + k0 + h * 8;
  v8bf lo = *(const v8bf*)p;
  v8bf hi = *(const v8bf*)(p + 16);
  v16bf r;
#pragma unroll
  for (int i = 0; i < 8; ++i) { r[i] = lo[i]; r[i + 8] = hi[i]; }
  return r;
}

// B-matrix fragment (32x16 bf16 tile): lane (n = l&15, half = l>>4) holds
// column col0+n, K = k0 + 8*half + {0..7} and k0 + 16 + 8*half + {0..7}. baseT is the
// transposed operand so column n is a contiguous row.
DEVINL v16bf load_b_frag(const bf16_t* baseT, int stride, int col0, int k0, int lane) {
  int n = lane & 15, h = lane >> 4;
  const bf16_t* p = baseT + (size_t)(col0 + n) * stride + k0 + h * 8;
  v8bf lo = *(const v8bf*)p;
  v8bf hi = *(const v8bf*)(p + 16);
  v16bf r;
#pragma unroll
  for (int i = 0; i < 8; ++i) { r[i] = lo[i]; r[i + 8] = hi[i]; }
  return r;
}

DEVINL v8f wmma_bf16(v16bf a, v16bf b, v8f c) {
  return __builtin_amdgcn_wmma_f32_16x16x32_bf16(false, a, false, b, (short)0, c,
                                                 false, false);
}

// ---------------- prep kernels ----------------

__global__ void k_scale_w2(const float* __restrict__ xc, const float* __restrict__ W2,
                           bf16_t* __restrict__ xcw2) {
  int idx = blockIdx.x * 256 + threadIdx.x;
  int d = idx & (D - 1);
  xcw2[idx] = f2bf(xc[idx] * W2[d]);
}

__global__ void k_s0s1(const float* __restrict__ xc, const float* __restrict__ xq,
                       const float* __restrict__ W0, const float* __restrict__ W1,
                       float* __restrict__ s0, float* __restrict__ s1) {
  int lane = threadIdx.x & 31;
  int w = threadIdx.x >> 5;
  int r = blockIdx.x * 8 + w;
  const float* row; const float* wv; float* out; int oi;
  if (r < B * C) { row = xc + (size_t)r * D; wv = W0; out = s0; oi = r; }
  else { int r2 = r - B * C; row = xq + (size_t)r2 * D; wv = W1; out = s1; oi = r2; }
  float acc = 0.f;
#pragma unroll
  for (int i = 0; i < 4; ++i) acc += row[lane + 32 * i] * wv[lane + 32 * i];
#pragma unroll
  for (int off = 16; off > 0; off >>= 1) acc += __shfl_xor(acc, off, 32);
  if (lane == 0) out[oi] = acc;
}

__global__ void k_xcT(const float* __restrict__ xc, bf16_t* __restrict__ xcT) {
  __shared__ float lds[64][65];
  int b = blockIdx.z;
  int c0 = blockIdx.x * 64, d0 = blockIdx.y * 64;
  int tx = threadIdx.x & 63, tz = threadIdx.x >> 6;
#pragma unroll
  for (int i = 0; i < 16; ++i) {
    int cl = tz + 4 * i;
    lds[cl][tx] = xc[((size_t)(b * C + c0 + cl)) * D + d0 + tx];
  }
  __syncthreads();
#pragma unroll
  for (int i = 0; i < 16; ++i) {
    int dl = tz + 4 * i;
    xcT[((size_t)(b * D + d0 + dl)) * C + c0 + tx] = f2bf(lds[tx][dl]);
  }
}

__global__ void k_xq(const float* __restrict__ xq, bf16_t* __restrict__ xqbf,
                     bf16_t* __restrict__ xqT) {
  __shared__ float lds[64][65];
  int b = blockIdx.z;
  int q0 = blockIdx.x * 64, d0 = blockIdx.y * 64;
  int tx = threadIdx.x & 63, tz = threadIdx.x >> 6;
#pragma unroll
  for (int i = 0; i < 16; ++i) {
    int ql = tz + 4 * i;
    float v = xq[((size_t)(b * Q + q0 + ql)) * D + d0 + tx];
    lds[ql][tx] = v;
    xqbf[((size_t)(b * Q + q0 + ql)) * D + d0 + tx] = f2bf(v);
  }
  __syncthreads();
#pragma unroll
  for (int i = 0; i < 16; ++i) {
    int dl = tz + 4 * i;
    xqT[((size_t)(b * D + d0 + dl)) * Q + q0 + tx] = f2bf(lds[tx][dl]);
  }
}

// ---------------- GEMM1: S = xcw2 @ xq^T + s0 + s1 + bias ----------------

__global__ void k_gemm_S(const bf16_t* __restrict__ xcw2, const bf16_t* __restrict__ xqbf,
                         const float* __restrict__ s0, const float* __restrict__ s1,
                         const float* __restrict__ biasp, float* __restrict__ S) {
  int b = blockIdx.z;
  int w = threadIdx.x >> 5, lane = threadIdx.x & 31;
  int c0 = blockIdx.x * 64 + (w >> 2) * 32;
  int q0 = blockIdx.y * 128 + (w & 3) * 32;
  const bf16_t* A = xcw2 + (size_t)b * C * D;
  const bf16_t* Bm = xqbf + (size_t)b * Q * D;
  v8f acc[2][2] = {};
#pragma unroll
  for (int k0 = 0; k0 < D; k0 += 32) {
    v16bf a0 = load_a_frag(A, D, c0, k0, lane);
    v16bf a1 = load_a_frag(A, D, c0 + 16, k0, lane);
    v16bf b0 = load_b_frag(Bm, D, q0, k0, lane);
    v16bf b1 = load_b_frag(Bm, D, q0 + 16, k0, lane);
    acc[0][0] = wmma_bf16(a0, b0, acc[0][0]);
    acc[0][1] = wmma_bf16(a0, b1, acc[0][1]);
    acc[1][0] = wmma_bf16(a1, b0, acc[1][0]);
    acc[1][1] = wmma_bf16(a1, b1, acc[1][1]);
  }
  float biasv = biasp[0];
  int nlo = lane & 15, h = lane >> 4;
  float s1v[2];
  s1v[0] = s1[b * Q + q0 + nlo];
  s1v[1] = s1[b * Q + q0 + 16 + nlo];
#pragma unroll
  for (int i = 0; i < 2; ++i) {
#pragma unroll
    for (int r = 0; r < 8; ++r) {
      int c = c0 + i * 16 + r + 8 * h;
      float s0v = s0[b * C + c] + biasv;
      size_t rowoff = ((size_t)(b * C + c)) * Q;
#pragma unroll
      for (int j = 0; j < 2; ++j)
        S[rowoff + q0 + j * 16 + nlo] = acc[i][j][r] + s0v + s1v[j];
    }
  }
}

// ---------------- softmax passes ----------------

__global__ void k_rowsoftmax(const float* __restrict__ S, const float* __restrict__ q_mask,
                             bf16_t* __restrict__ Srow) {
  int lane = threadIdx.x & 31, w = threadIdx.x >> 5;
  int r = blockIdx.x * 8 + w;  // [0, B*C)
  int b = r >> 10;             // C = 1024
  const float* row = S + (size_t)r * Q;
  const float* qm = q_mask + b * Q;
  float v[8];
  float m = -3.4e38f;
#pragma unroll
  for (int i = 0; i < 8; ++i) {
    int q = lane + 32 * i;
    v[i] = row[q] + NEGV * (1.0f - qm[q]);
    m = fmaxf(m, v[i]);
  }
#pragma unroll
  for (int off = 16; off > 0; off >>= 1) m = fmaxf(m, __shfl_xor(m, off, 32));
  float s = 0.f;
#pragma unroll
  for (int i = 0; i < 8; ++i) { v[i] = __expf(v[i] - m); s += v[i]; }
#pragma unroll
  for (int off = 16; off > 0; off >>= 1) s += __shfl_xor(s, off, 32);
  float inv = 1.0f / s;
  bf16_t* orow = Srow + (size_t)r * Q;
#pragma unroll
  for (int i = 0; i < 8; ++i) orow[lane + 32 * i] = f2bf(v[i] * inv);
}

__global__ void k_colstats(const float* __restrict__ S, const float* __restrict__ c_mask,
                           float* __restrict__ pmax, float* __restrict__ psum) {
  int q = threadIdx.x;
  int chunk = blockIdx.x & 3;
  int b = blockIdx.x >> 2;
  int cbase = chunk * (C / 4);
  const float* Sb = S + ((size_t)b * C) * Q;
  const float* cm = c_mask + b * C;
  float m = -3.4e38f, s = 0.f;
  for (int i = 0; i < C / 4; ++i) {
    int c = cbase + i;
    float v = Sb[(size_t)c * Q + q] + NEGV * (1.0f - cm[c]);
    float nm = fmaxf(m, v);
    s = s * __expf(m - nm) + __expf(v - nm);
    m = nm;
  }
  pmax[blockIdx.x * Q + q] = m;
  psum[blockIdx.x * Q + q] = s;
}

__global__ void k_colcombine(const float* __restrict__ pmax, const float* __restrict__ psum,
                             float* __restrict__ colmax, float* __restrict__ colinv) {
  int idx = blockIdx.x * 256 + threadIdx.x;  // [0, B*Q)
  int b = idx >> 8, q = idx & 255;
  float m = -3.4e38f;
#pragma unroll
  for (int k = 0; k < 4; ++k) m = fmaxf(m, pmax[(b * 4 + k) * Q + q]);
  float s = 0.f;
#pragma unroll
  for (int k = 0; k < 4; ++k)
    s += psum[(b * 4 + k) * Q + q] * __expf(pmax[(b * 4 + k) * Q + q] - m);
  colmax[idx] = m;
  colinv[idx] = 1.0f / s;
}

__global__ void k_buildST(const float* __restrict__ S, const float* __restrict__ colmax,
                          const float* __restrict__ colinv, bf16_t* __restrict__ ST) {
  __shared__ float lds[64][65];
  int b = blockIdx.z;
  int c0 = blockIdx.x * 64, q0 = blockIdx.y * 64;
  int tx = threadIdx.x & 63, tz = threadIdx.x >> 6;
  float cmv = colmax[b * Q + q0 + tx];
  float civ = colinv[b * Q + q0 + tx];
#pragma unroll
  for (int i = 0; i < 16; ++i) {
    int cl = tz + 4 * i;
    float v = S[((size_t)(b * C + c0 + cl)) * Q + q0 + tx];
    lds[cl][tx] = __expf(v - cmv) * civ;
  }
  __syncthreads();
#pragma unroll
  for (int i = 0; i < 16; ++i) {
    int ql = tz + 4 * i;
    ST[((size_t)(b * Q + q0 + ql)) * C + c0 + tx] = f2bf(lds[tx][ql]);
  }
}

// ---------------- GEMM2: TT = (S_T @ xc)^T ----------------

__global__ void k_gemm_T(const bf16_t* __restrict__ ST, const bf16_t* __restrict__ xcT,
                         bf16_t* __restrict__ TT) {
  int b = blockIdx.z;
  int w = threadIdx.x >> 5, lane = threadIdx.x & 31;
  int q0 = blockIdx.x * 128 + (w >> 1) * 32;
  int d0 = blockIdx.y * 64 + (w & 1) * 32;
  const bf16_t* A = ST + (size_t)b * Q * C;    // Q x C, stride C
  const bf16_t* Bt = xcT + (size_t)b * D * C;  // D x C, row d = xc column d
  v8f acc[2][2] = {};
#pragma unroll 4
  for (int k0 = 0; k0 < C; k0 += 32) {
    v16bf a0 = load_a_frag(A, C, q0, k0, lane);
    v16bf a1 = load_a_frag(A, C, q0 + 16, k0, lane);
    v16bf b0 = load_b_frag(Bt, C, d0, k0, lane);
    v16bf b1 = load_b_frag(Bt, C, d0 + 16, k0, lane);
    acc[0][0] = wmma_bf16(a0, b0, acc[0][0]);
    acc[0][1] = wmma_bf16(a0, b1, acc[0][1]);
    acc[1][0] = wmma_bf16(a1, b0, acc[1][0]);
    acc[1][1] = wmma_bf16(a1, b1, acc[1][1]);
  }
  int nlo = lane & 15, h = lane >> 4;
#pragma unroll
  for (int i = 0; i < 2; ++i)
#pragma unroll
    for (int r = 0; r < 8; ++r) {
      int q = q0 + i * 16 + r + 8 * h;
#pragma unroll
      for (int j = 0; j < 2; ++j) {
        int d = d0 + j * 16 + nlo;
        TT[((size_t)(b * D + d)) * Q + q] = f2bf(acc[i][j][r]);
      }
    }
}

// ---- GEMM3+4 fused: c2q = S_@xq, q2c = S_@T, + concat epilogue ----

__global__ void k_gemm_out(const bf16_t* __restrict__ Srow, const bf16_t* __restrict__ xqT,
                           const bf16_t* __restrict__ TT, const float* __restrict__ xc,
                           float* __restrict__ out) {
  int b = blockIdx.y;
  int w = threadIdx.x >> 5, lane = threadIdx.x & 31;
  int c0 = blockIdx.x * 64 + (w >> 2) * 32;
  int d0 = (w & 3) * 32;
  const bf16_t* A = Srow + (size_t)b * C * Q;  // C x Q, stride Q
  const bf16_t* Bq = xqT + (size_t)b * D * Q;  // D x Q
  const bf16_t* Bt = TT + (size_t)b * D * Q;   // D x Q
  v8f acq[2][2] = {}, aqc[2][2] = {};
#pragma unroll 2
  for (int k0 = 0; k0 < Q; k0 += 32) {
    v16bf a0 = load_a_frag(A, Q, c0, k0, lane);
    v16bf a1 = load_a_frag(A, Q, c0 + 16, k0, lane);
    v16bf bq0 = load_b_frag(Bq, Q, d0, k0, lane);
    v16bf bq1 = load_b_frag(Bq, Q, d0 + 16, k0, lane);
    v16bf bt0 = load_b_frag(Bt, Q, d0, k0, lane);
    v16bf bt1 = load_b_frag(Bt, Q, d0 + 16, k0, lane);
    acq[0][0] = wmma_bf16(a0, bq0, acq[0][0]);
    acq[0][1] = wmma_bf16(a0, bq1, acq[0][1]);
    acq[1][0] = wmma_bf16(a1, bq0, acq[1][0]);
    acq[1][1] = wmma_bf16(a1, bq1, acq[1][1]);
    aqc[0][0] = wmma_bf16(a0, bt0, aqc[0][0]);
    aqc[0][1] = wmma_bf16(a0, bt1, aqc[0][1]);
    aqc[1][0] = wmma_bf16(a1, bt0, aqc[1][0]);
    aqc[1][1] = wmma_bf16(a1, bt1, aqc[1][1]);
  }
  int nlo = lane & 15, h = lane >> 4;
#pragma unroll
  for (int i = 0; i < 2; ++i)
#pragma unroll
    for (int r = 0; r < 8; ++r) {
      int c = c0 + i * 16 + r + 8 * h;
      size_t inrow = ((size_t)(b * C + c)) * D;
      size_t outrow = ((size_t)(b * C + c)) * (4 * D);
#pragma unroll
      for (int j = 0; j < 2; ++j) {
        int d = d0 + j * 16 + nlo;
        float x = xc[inrow + d];
        float cq = acq[i][j][r];
        float qc = aqc[i][j][r];
        out[outrow + d] = x;
        out[outrow + D + d] = cq;
        out[outrow + 2 * D + d] = x * cq;
        out[outrow + 3 * D + d] = x * qc;
      }
    }
}

extern "C" void kernel_launch(void* const* d_in, const int* in_sizes, int n_in,
                              void* d_out, int out_size, void* d_ws, size_t ws_size,
                              hipStream_t stream) {
  (void)in_sizes; (void)n_in; (void)out_size; (void)ws_size;
  const float* x_cont = (const float*)d_in[0];
  const float* x_ques = (const float*)d_in[1];
  const float* c_mask = (const float*)d_in[2];
  const float* q_mask = (const float*)d_in[3];
  const float* W0 = (const float*)d_in[4];
  const float* W1 = (const float*)d_in[5];
  const float* W2 = (const float*)d_in[6];
  const float* bias = (const float*)d_in[7];
  float* out = (float*)d_out;

  char* w = (char*)d_ws;
  auto alloc = [&](size_t bytes) {
    char* p = w;
    w += (bytes + 255) & ~(size_t)255;
    return p;
  };
  float* S      = (float*)alloc((size_t)B * C * Q * 4);
  bf16_t* Srow  = (bf16_t*)alloc((size_t)B * C * Q * 2);
  bf16_t* ST    = (bf16_t*)alloc((size_t)B * Q * C * 2);
  bf16_t* xcw2  = (bf16_t*)alloc((size_t)B * C * D * 2);
  bf16_t* xcT   = (bf16_t*)alloc((size_t)B * D * C * 2);
  bf16_t* xqbf  = (bf16_t*)alloc((size_t)B * Q * D * 2);
  bf16_t* xqT   = (bf16_t*)alloc((size_t)B * D * Q * 2);
  bf16_t* TT    = (bf16_t*)alloc((size_t)B * D * Q * 2);
  float* s0     = (float*)alloc((size_t)B * C * 4);
  float* s1     = (float*)alloc((size_t)B * Q * 4);
  float* pmax   = (float*)alloc((size_t)B * 4 * Q * 4);
  float* psum   = (float*)alloc((size_t)B * 4 * Q * 4);
  float* colmax = (float*)alloc((size_t)B * Q * 4);
  float* colinv = (float*)alloc((size_t)B * Q * 4);

  k_scale_w2<<<(B * C * D) / 256, 256, 0, stream>>>(x_cont, W2, xcw2);
  k_s0s1<<<(B * C + B * Q) / 8, 256, 0, stream>>>(x_cont, x_ques, W0, W1, s0, s1);
  k_xcT<<<dim3(C / 64, D / 64, B), 256, 0, stream>>>(x_cont, xcT);
  k_xq<<<dim3(Q / 64, D / 64, B), 256, 0, stream>>>(x_ques, xqbf, xqT);
  k_gemm_S<<<dim3(C / 64, Q / 128, B), 256, 0, stream>>>(xcw2, xqbf, s0, s1, bias, S);
  k_rowsoftmax<<<(B * C) / 8, 256, 0, stream>>>(S, q_mask, Srow);
  k_colstats<<<B * 4, 256, 0, stream>>>(S, c_mask, pmax, psum);
  k_colcombine<<<(B * Q) / 256, 256, 0, stream>>>(pmax, psum, colmax, colinv);
  k_buildST<<<dim3(C / 64, Q / 64, B), 256, 0, stream>>>(S, colmax, colinv, ST);
  k_gemm_T<<<dim3(Q / 128, D / 64, B), 256, 0, stream>>>(ST, xcT, TT);
  k_gemm_out<<<dim3(C / 64, B), 256, 0, stream>>>(Srow, xqT, TT, x_cont, out);
}
